// CrossSRA_12884901888495
// MI455X (gfx1250) — matrix operand order repaired and checked by reading, not compiled, not run on hardware
//
#include <hip/hip_runtime.h>
#include <hip/hip_bf16.h>

typedef __bf16 bf16;
typedef __attribute__((ext_vector_type(16))) __bf16 v16bf;
typedef __attribute__((ext_vector_type(8)))  float  v8f;

#define WMMA_BF16(a, b, c) \
  __builtin_amdgcn_wmma_f32_16x16x32_bf16(false, (a), false, (b), (short)0, (c), false, false)

// --------------------------------------------------------------------------
// CDNA5 async global->LDS copy (ASYNCcnt path), guarded so we fall back to a
// plain VGPR round-trip if this toolchain doesn't declare the builtin.
// Probe feedback: parameters are pointers to 16-byte int vectors, src in the
// global address space, dst in LDS.
// --------------------------------------------------------------------------
#if defined(__HIP_DEVICE_COMPILE__) && \
    __has_builtin(__builtin_amdgcn_global_load_async_to_lds_b128)
#define USE_ASYNC_LDS 1
#else
#define USE_ASYNC_LDS 0
#endif

#if USE_ASYNC_LDS
typedef int v4i __attribute__((__vector_size__(4 * sizeof(int))));
typedef __attribute__((address_space(1))) v4i* gv4i_p;
typedef __attribute__((address_space(3))) v4i* lv4i_p;
__device__ __forceinline__ gv4i_p as_global_v4(const void* p) {
  return (gv4i_p)(unsigned long long)p;
}
__device__ __forceinline__ lv4i_p as_lds_v4(void* p) {
  // generic LDS pointers keep the LDS offset in the low 32 bits
  return (lv4i_p)(unsigned int)(unsigned long long)p;
}
#if __has_builtin(__builtin_amdgcn_s_wait_asynccnt)
#define WAIT_ASYNC() __builtin_amdgcn_s_wait_asynccnt(0)
#else
#define WAIT_ASYNC() asm volatile("s_wait_asynccnt 0" ::: "memory")
#endif
#endif

// ---------------------------------------------------------------------------
// Fragment loaders from LDS per CDNA5 WMMA VGPR layouts (CDNA5 ISA)
// A 16x32 bf16: lanes 0-15 row M=lane, K = [0..7],[16..23]; lanes 16-31 K=+8.
// B 32x16 bf16: lane holds col N=lane%16, K = [0..7],[16..23]; lanes 16-31 K=+8.
// C 16x16 f32 : row = (lane/16)*8 + vgpr, col = lane%16.
// ---------------------------------------------------------------------------
__device__ __forceinline__ v16bf frag_a_lds(const bf16* tile, int lda, int row0,
                                            int k0, int lane) {
  int r  = row0 + (lane & 15);
  int kb = k0 + ((lane >> 4) << 3);
  const bf16* p = tile + r * lda + kb;
  v16bf f;
#pragma unroll
  for (int i = 0; i < 8; ++i) { f[i] = p[i]; f[i + 8] = p[i + 16]; }
  return f;
}

__device__ __forceinline__ v16bf frag_b_lds(const bf16* tile, int ldb, int n0,
                                            int k0, int lane) {
  int n  = n0 + (lane & 15);
  int kb = k0 + ((lane >> 4) << 3);
  const bf16* p = tile + n * ldb + kb;
  v16bf f;
#pragma unroll
  for (int i = 0; i < 8; ++i) { f[i] = p[i]; f[i + 8] = p[i + 16]; }
  return f;
}

// ---------------------------------------------------------------------------
// 16-element row stagers (16B-aligned by construction -> b128 traffic)
// ---------------------------------------------------------------------------
__device__ __forceinline__ void stage16(bf16* dst, const float* src) {
  const float4* s4 = (const float4*)src;
#pragma unroll
  for (int i = 0; i < 4; ++i) {
    float4 t = s4[i];
    dst[4 * i + 0] = (bf16)t.x;
    dst[4 * i + 1] = (bf16)t.y;
    dst[4 * i + 2] = (bf16)t.z;
    dst[4 * i + 3] = (bf16)t.w;
  }
}
__device__ __forceinline__ void stage16(bf16* dst, const bf16* src) {
  const uint4* s4 = (const uint4*)src;
  uint4 t0 = s4[0], t1 = s4[1];
  *(uint4*)(dst)     = t0;
  *(uint4*)(dst + 8) = t1;
}

// ---------------------------------------------------------------------------
// Generic GEMM: Out[m,n] = sum_k X[m,k]*W[n,k] + bias[n]
// 128x128 block tile, BK=32, 256 threads = 8 waves (2x4), wave tile 64x32.
// Double-buffered LDS: next tile's global loads overlap current tile's WMMAs.
// fp32 (or bf16) -> bf16 conversion happens while staging into LDS.
// ---------------------------------------------------------------------------
template <typename InT, typename OutT>
__global__ __launch_bounds__(256) void gemm_xwT(const InT* __restrict__ X,
                                                const float* __restrict__ W,
                                                const float* __restrict__ bias,
                                                OutT* __restrict__ Out,
                                                int M, int N, int K) {
  __shared__ bf16 As[2][128][40];
  __shared__ bf16 Bs[2][128][40];
  const int tid  = threadIdx.x;
  const int lane = tid & 31;
  const int wave = tid >> 5;
  const int wm   = wave >> 2;  // 0..1
  const int wn   = wave & 3;   // 0..3
  const int gm0  = blockIdx.y * 128;
  const int gn0  = blockIdx.x * 128;

  v8f acc[4][2] = {};

  const int srow = tid >> 1;
  const int skb  = (tid & 1) << 4;
  const InT*   aptr = X + (size_t)(gm0 + srow) * K + skb;
  const float* bptr = W + (size_t)(gn0 + srow) * K + skb;

  stage16(&As[0][srow][skb], aptr);
  stage16(&Bs[0][srow][skb], bptr);
  __syncthreads();

  const int nk = K >> 5;
  for (int t = 0; t < nk; ++t) {
    const int cur = t & 1;
    if (t + 1 < nk) {
      stage16(&As[cur ^ 1][srow][skb], aptr + (size_t)(t + 1) * 32);
      stage16(&Bs[cur ^ 1][srow][skb], bptr + (size_t)(t + 1) * 32);
    }
    if (t + 2 < nk) {
      __builtin_prefetch(aptr + (size_t)(t + 2) * 32, 0, 3);
      __builtin_prefetch(bptr + (size_t)(t + 2) * 32, 0, 3);
    }
    v16bf bf0 = frag_b_lds(&Bs[cur][0][0], 40, wn * 32, 0, lane);
    v16bf bf1 = frag_b_lds(&Bs[cur][0][0], 40, wn * 32 + 16, 0, lane);
#pragma unroll
    for (int i = 0; i < 4; ++i) {
      v16bf af = frag_a_lds(&As[cur][0][0], 40, wm * 64 + i * 16, 0, lane);
      acc[i][0] = WMMA_BF16(af, bf0, acc[i][0]);
      acc[i][1] = WMMA_BF16(af, bf1, acc[i][1]);
    }
    __syncthreads();
  }

#pragma unroll
  for (int i = 0; i < 4; ++i) {
    int gr = gm0 + wm * 64 + i * 16 + ((lane >> 4) << 3);
#pragma unroll
    for (int j = 0; j < 2; ++j) {
      int gc = gn0 + wn * 32 + j * 16 + (lane & 15);
      float bv = bias[gc];
#pragma unroll
      for (int r = 0; r < 8; ++r)
        Out[(size_t)(gr + r) * N + gc] = (OutT)(acc[i][j][r] + bv);
    }
  }
}

// ---------------------------------------------------------------------------
// Conv 8x8 stride 8 as implicit-im2col GEMM (double buffered).
// M=512 rows (b*64 + i*8+j), K=49152 (c*64 + dy*8 + dx), N=768.
// conv_w is already [O, K] row-major for this k ordering.
// ---------------------------------------------------------------------------
__device__ __forceinline__ void stage_conv_a(bf16* dst, const float* kvx,
                                             int bb, int pi, int pj, int kbase) {
#pragma unroll
  for (int e = 0; e < 16; ++e) {
    int k   = kbase + e;
    int c   = k >> 6;
    int rem = k & 63;
    int dy  = rem >> 3;
    int dx  = rem & 7;
    int pos = (pi * 8 + dy) * 64 + (pj * 8 + dx);
    dst[e] = (bf16)kvx[((size_t)bb * 4096 + pos) * 768 + c];
  }
}

__global__ __launch_bounds__(256) void gemm_conv(const float* __restrict__ kvx,
                                                 const float* __restrict__ Wc,
                                                 const float* __restrict__ bias,
                                                 float* __restrict__ Out) {
  __shared__ bf16 As[2][128][40];
  __shared__ bf16 Bs[2][128][40];
  const int tid  = threadIdx.x;
  const int lane = tid & 31;
  const int wave = tid >> 5;
  const int wm   = wave >> 2;
  const int wn   = wave & 3;
  const int gm0  = blockIdx.y * 128;
  const int gn0  = blockIdx.x * 128;

  v8f acc[4][2] = {};

  const int rloc = tid >> 1;
  const int skb  = (tid & 1) << 4;
  const int r    = gm0 + rloc;
  const int bb   = r >> 6;
  const int pp   = r & 63;
  const int pi   = pp >> 3;
  const int pj   = pp & 7;
  const float* bptr = Wc + (size_t)(gn0 + rloc) * 49152 + skb;

  stage_conv_a(&As[0][rloc][skb], kvx, bb, pi, pj, skb);
  stage16(&Bs[0][rloc][skb], bptr);
  __syncthreads();

  const int nk = 49152 >> 5;
  for (int t = 0; t < nk; ++t) {
    const int cur = t & 1;
    if (t + 1 < nk) {
      stage_conv_a(&As[cur ^ 1][rloc][skb], kvx, bb, pi, pj, (t + 1) * 32 + skb);
      stage16(&Bs[cur ^ 1][rloc][skb], bptr + (size_t)(t + 1) * 32);
    }
    if (t + 2 < nk) __builtin_prefetch(bptr + (size_t)(t + 2) * 32, 0, 3);

    v16bf bf0 = frag_b_lds(&Bs[cur][0][0], 40, wn * 32, 0, lane);
    v16bf bf1 = frag_b_lds(&Bs[cur][0][0], 40, wn * 32 + 16, 0, lane);
#pragma unroll
    for (int i = 0; i < 4; ++i) {
      v16bf af = frag_a_lds(&As[cur][0][0], 40, wm * 64 + i * 16, 0, lane);
      acc[i][0] = WMMA_BF16(af, bf0, acc[i][0]);
      acc[i][1] = WMMA_BF16(af, bf1, acc[i][1]);
    }
    __syncthreads();
  }

#pragma unroll
  for (int i = 0; i < 4; ++i) {
    int gr = gm0 + wm * 64 + i * 16 + ((lane >> 4) << 3);
#pragma unroll
    for (int j = 0; j < 2; ++j) {
      int gc = gn0 + wn * 32 + j * 16 + (lane & 15);
      float bv = bias[gc];
#pragma unroll
      for (int rr = 0; rr < 8; ++rr)
        Out[(size_t)(gr + rr) * 768 + gc] = acc[i][j][rr] + bv;
    }
  }
}

// ---------------------------------------------------------------------------
// LayerNorm over rows of 768
// ---------------------------------------------------------------------------
__global__ __launch_bounds__(256) void layernorm_k(const float* __restrict__ x,
                                                   const float* __restrict__ g,
                                                   const float* __restrict__ b,
                                                   float* __restrict__ out) {
  __shared__ float red[256];
  const int row = blockIdx.x;
  const float* xr = x + (size_t)row * 768;
  float s = 0.f;
  for (int i = threadIdx.x; i < 768; i += 256) s += xr[i];
  red[threadIdx.x] = s;
  __syncthreads();
  for (int o = 128; o > 0; o >>= 1) {
    if (threadIdx.x < o) red[threadIdx.x] += red[threadIdx.x + o];
    __syncthreads();
  }
  float mu = red[0] / 768.f;
  __syncthreads();
  float v = 0.f;
  for (int i = threadIdx.x; i < 768; i += 256) {
    float d = xr[i] - mu;
    v += d * d;
  }
  red[threadIdx.x] = v;
  __syncthreads();
  for (int o = 128; o > 0; o >>= 1) {
    if (threadIdx.x < o) red[threadIdx.x] += red[threadIdx.x + o];
    __syncthreads();
  }
  float inv = rsqrtf(red[0] / 768.f + 1e-5f);
  for (int i = threadIdx.x; i < 768; i += 256)
    out[(size_t)row * 768 + i] = (xr[i] - mu) * inv * g[i] + b[i];
}

// ---------------------------------------------------------------------------
// Bilinear resize 64x64 -> 8x8 (align_corners=False): scale 8 => all taps 0.5
// ---------------------------------------------------------------------------
__global__ void resize_bias_k(const float* __restrict__ kvb,
                              float* __restrict__ bias8) {
  int t = blockIdx.x * blockDim.x + threadIdx.x;
  if (t >= 512) return;
  int b = t >> 6, p = t & 63, i = p >> 3, j = p & 7;
  int y0 = 8 * i + 3, y1 = 8 * i + 4, x0 = 8 * j + 3, x1 = 8 * j + 4;
  const float* src = kvb + (size_t)b * 64 * 64;
  bias8[t] = 0.25f * (src[y0 * 64 + x0] + src[y0 * 64 + x1] +
                      src[y1 * 64 + x0] + src[y1 * 64 + x1]);
}

// ---------------------------------------------------------------------------
// Fused attention: per (b, h) and 64-row q slab.
// S = Q K^T (WMMA, dh=96), scale + resized bias + softmax (M=64 per thread),
// O = P V (WMMA, K=64). 128 threads = 4 waves, wave owns 16 q rows.
// Q/K staged via GLOBAL_LOAD_ASYNC_TO_LDS_B128 when available (pure bf16
// copies of aligned 16B chunks); Ss overlays dead Qs/Ks region.
// ---------------------------------------------------------------------------
__global__ __launch_bounds__(128) void attn_k(const bf16* __restrict__ q,
                                              const bf16* __restrict__ k,
                                              const bf16* __restrict__ v,
                                              const float* __restrict__ bias8,
                                              bf16* __restrict__ xout) {
  // byte layout (bf16 pitches 16B multiples):
  //   Qs  [64][104] bf16 @ 0       (13312 B)   dead after QK^T
  //   Ks  [64][104] bf16 @ 13312   (13312 B)   dead after QK^T
  //   Vts [96][72]  bf16 @ 26624   (13824 B)
  //   Ss  [64][66]  f32  @ 0       (16896 B)   overlays Qs/Ks
  //   Ps  [64][72]  bf16 @ 40448   ( 9216 B)   total 49664 B
  __shared__ __align__(16) unsigned char smem[49664];
  bf16 (*Qs)[104] = (bf16(*)[104])(smem);
  bf16 (*Ks)[104] = (bf16(*)[104])(smem + 13312);
  bf16 (*Vts)[72] = (bf16(*)[72])(smem + 26624);
  float (*Ss)[66] = (float(*)[66])(smem);
  bf16 (*Ps)[72]  = (bf16(*)[72])(smem + 40448);

  const int tid  = threadIdx.x;
  const int lane = tid & 31;
  const int w    = tid >> 5;    // wave 0..3
  const int bh   = blockIdx.y;  // b*8 + h
  const int b    = bh >> 3;
  const int hh   = bh & 7;
  const int q0   = blockIdx.x * 64;

#if USE_ASYNC_LDS
  // 64 rows x 96 bf16 = 12 x 16B chunks per row, for Q and K each.
  for (int idx = tid; idx < 768; idx += 128) {
    int r = idx / 12, c8 = (idx % 12) * 8;
    const bf16* gq = q + ((size_t)b * 4096 + q0 + r) * 768 + hh * 96 + c8;
    const bf16* gk = k + ((size_t)b * 64 + r) * 768 + hh * 96 + c8;
    __builtin_amdgcn_global_load_async_to_lds_b128(as_global_v4(gq),
                                                   as_lds_v4(&Qs[r][c8]), 0, 0);
    __builtin_amdgcn_global_load_async_to_lds_b128(as_global_v4(gk),
                                                   as_lds_v4(&Ks[r][c8]), 0, 0);
  }
  for (int idx = tid; idx < 64 * 96; idx += 128) {
    int r = idx / 96, d = idx % 96;
    Vts[d][r] = v[((size_t)b * 64 + r) * 768 + hh * 96 + d];
  }
  WAIT_ASYNC();
#else
  for (int idx = tid; idx < 64 * 96; idx += 128) {
    int r = idx / 96, d = idx % 96;
    Qs[r][d]  = q[((size_t)b * 4096 + q0 + r) * 768 + hh * 96 + d];
    Ks[r][d]  = k[((size_t)b * 64 + r) * 768 + hh * 96 + d];
    Vts[d][r] = v[((size_t)b * 64 + r) * 768 + hh * 96 + d];
  }
#endif
  __syncthreads();

  // S = Q K^T : wave rows 16w..16w+15, cols 0..63
  v8f accS[4] = {};
  for (int k0 = 0; k0 < 96; k0 += 32) {
    v16bf af = frag_a_lds(&Qs[0][0], 104, 16 * w, k0, lane);
#pragma unroll
    for (int j = 0; j < 4; ++j) {
      v16bf bf = frag_b_lds(&Ks[0][0], 104, 16 * j, k0, lane);
      accS[j] = WMMA_BF16(af, bf, accS[j]);
    }
  }
  __syncthreads();  // Qs/Ks fully consumed; Ss may overwrite them now
#pragma unroll
  for (int j = 0; j < 4; ++j) {
    int col = 16 * j + (lane & 15);
    int row = 16 * w + ((lane >> 4) << 3);
#pragma unroll
    for (int r = 0; r < 8; ++r) Ss[row + r][col] = accS[j][r];
  }
  __syncthreads();

  // softmax over M=64 (one row per thread, threads 0..63)
  if (tid < 64) {
    const float scale = 0.10206207261596575f;  // 96^-0.5
    float vals[64];
    float mx = -1e30f;
#pragma unroll
    for (int m = 0; m < 64; ++m) {
      float t = Ss[tid][m] * scale + bias8[b * 64 + m];
      vals[m] = t;
      mx = fmaxf(mx, t);
    }
    float sum = 0.f;
#pragma unroll
    for (int m = 0; m < 64; ++m) {
      float e = __expf(vals[m] - mx);
      vals[m] = e;
      sum += e;
    }
    float inv = 1.0f / sum;
#pragma unroll
    for (int m = 0; m < 64; ++m) Ps[tid][m] = (bf16)(vals[m] * inv);
  }
  __syncthreads();

  // O = P V : wave rows 16w..16w+15, cols 0..95
  v8f accO[6] = {};
  for (int k0 = 0; k0 < 64; k0 += 32) {
    v16bf af = frag_a_lds(&Ps[0][0], 72, 16 * w, k0, lane);
#pragma unroll
    for (int j = 0; j < 6; ++j) {
      v16bf bf = frag_b_lds(&Vts[0][0], 72, 16 * j, k0, lane);
      accO[j] = WMMA_BF16(af, bf, accO[j]);
    }
  }
#pragma unroll
  for (int j = 0; j < 6; ++j) {
    int col = 16 * j + (lane & 15);
    int row = 16 * w + ((lane >> 4) << 3);
#pragma unroll
    for (int r = 0; r < 8; ++r)
      xout[((size_t)b * 4096 + q0 + row + r) * 768 + hh * 96 + col] =
          (bf16)accO[j][r];
  }
}

// ---------------------------------------------------------------------------
extern "C" void kernel_launch(void* const* d_in, const int* in_sizes, int n_in,
                              void* d_out, int out_size, void* d_ws,
                              size_t ws_size, hipStream_t stream) {
  (void)in_sizes; (void)n_in; (void)out_size; (void)ws_size;
  const float* qx     = (const float*)d_in[0];
  const float* kvx    = (const float*)d_in[1];
  const float* kvb    = (const float*)d_in[2];
  const float* q_w    = (const float*)d_in[5];
  const float* q_b    = (const float*)d_in[6];
  const float* k_w    = (const float*)d_in[7];
  const float* k_b    = (const float*)d_in[8];
  const float* v_w    = (const float*)d_in[9];
  const float* v_b    = (const float*)d_in[10];
  const float* proj_w = (const float*)d_in[11];
  const float* proj_b = (const float*)d_in[12];
  const float* conv_w = (const float*)d_in[13];
  const float* conv_b = (const float*)d_in[14];
  const float* ln_w   = (const float*)d_in[15];
  const float* ln_b   = (const float*)d_in[16];
  float* out = (float*)d_out;

  char* ws = (char*)d_ws;
  size_t off = 0;
  auto alloc = [&](size_t bytes) -> char* {
    char* p = ws + off;
    off += (bytes + 255) & ~(size_t)255;
    return p;
  };
  bf16*  q_bf   = (bf16*)alloc((size_t)32768 * 768 * 2);
  bf16*  x_bf   = (bf16*)alloc((size_t)32768 * 768 * 2);
  bf16*  k_bf   = (bf16*)alloc((size_t)512 * 768 * 2);
  bf16*  v_bf   = (bf16*)alloc((size_t)512 * 768 * 2);
  float* conv_o = (float*)alloc((size_t)512 * 768 * 4);
  float* kv_ln  = (float*)alloc((size_t)512 * 768 * 4);
  float* bias8  = (float*)alloc((size_t)512 * 4);

  resize_bias_k<<<1, 512, 0, stream>>>(kvb, bias8);
  gemm_conv<<<dim3(6, 4), 256, 0, stream>>>(kvx, conv_w, conv_b, conv_o);
  layernorm_k<<<512, 256, 0, stream>>>(conv_o, ln_w, ln_b, kv_ln);
  gemm_xwT<float, bf16><<<dim3(6, 4), 256, 0, stream>>>(kv_ln, k_w, k_b, k_bf,
                                                        512, 768, 768);
  gemm_xwT<float, bf16><<<dim3(6, 4), 256, 0, stream>>>(kv_ln, v_w, v_b, v_bf,
                                                        512, 768, 768);
  gemm_xwT<float, bf16><<<dim3(6, 256), 256, 0, stream>>>(qx, q_w, q_b, q_bf,
                                                          32768, 768, 768);
  attn_k<<<dim3(64, 64), 128, 0, stream>>>(q_bf, k_bf, v_bf, bias8, x_bf);
  gemm_xwT<bf16, float><<<dim3(6, 256), 256, 0, stream>>>(x_bf, proj_w, proj_b,
                                                          out, 32768, 768, 768);
}
